// KVCacheAttention_21895743275138
// MI455X (gfx1250) — matrix operand order repaired and checked by reading, not compiled, not run on hardware
//
#include <hip/hip_runtime.h>

typedef __attribute__((ext_vector_type(16))) __bf16 v16bf;
typedef __attribute__((ext_vector_type(8)))  __bf16 v8bf;
typedef __attribute__((ext_vector_type(8)))  float  v8f;
typedef __attribute__((address_space(3)))    __bf16 lds_bf16;

#define B_SZ   2
#define SQ_SZ  2048
#define SC_SZ  2048
#define SKV_SZ 4096
#define E_SZ   1024
#define H_SZ   16
#define D_SZ   64

// ---------------------------------------------------------------------------
// WMMA helpers (CDNA5 v_wmma_f32_16x16x32_bf16, wave32)
// ---------------------------------------------------------------------------
static __device__ __forceinline__ v8f wmma_bf16(v16bf a, v16bf b, v8f c) {
  return __builtin_amdgcn_wmma_f32_16x16x32_bf16(false, a, false, b, (short)0, c,
                                                 false, false);
}

// A-operand 16x32 bf16 fragment: lane holds row (lane&15); elements 0..7 at
// K = kb, elements 8..15 at K = kb+16, kb = (lane<16 ? 0 : 8).
// Caller passes p = row_base + kbA (two aligned 16-byte chunks).
static __device__ __forceinline__ v16bf fragA_ld(const __bf16* p) {
  v8bf lo = *(const v8bf*)(p);
  v8bf hi = *(const v8bf*)(p + 16);
  v16bf o;
#pragma unroll
  for (int i = 0; i < 8; ++i) { o[i] = lo[i]; o[i + 8] = hi[i]; }
  return o;
}

// B-operand 32x16 bf16 fragment: lane holds column (lane&15); elements 0..7 at
// K = kb, elements 8..15 at K = kb+16, kb = (lane<16 ? 0 : 8).
static __device__ __forceinline__ v16bf fragB_ld(const __bf16* p) {
  v8bf lo = *(const v8bf*)(p);
  v8bf hi = *(const v8bf*)(p + 16);
  v16bf o;
#pragma unroll
  for (int i = 0; i < 8; ++i) { o[i] = lo[i]; o[i + 8] = hi[i]; }
  return o;
}

// Async global->LDS copy of 16 bytes per lane (CDNA5, ASYNCcnt-tracked).
static __device__ __forceinline__ void async_ld_b128(lds_bf16* l,
                                                     const __bf16* g) {
  asm volatile("global_load_async_to_lds_b128 %0, %1, off"
               :
               : "v"(l), "v"((unsigned long long)(uintptr_t)g)
               : "memory");
}

// ---------------------------------------------------------------------------
// fp32 -> bf16 convert pass
// ---------------------------------------------------------------------------
__global__ void cvt_bf16_kernel(const float* __restrict__ src,
                                __bf16* __restrict__ dst, int n) {
  int i = (blockIdx.x * 256 + threadIdx.x) * 4;
  if (i + 3 < n) {
    float4 v = *(const float4*)(src + i);
    dst[i + 0] = (__bf16)v.x;
    dst[i + 1] = (__bf16)v.y;
    dst[i + 2] = (__bf16)v.z;
    dst[i + 3] = (__bf16)v.w;
  } else {
    for (; i < n; ++i) dst[i] = (__bf16)src[i];
  }
}

// ---------------------------------------------------------------------------
// QKV projection: out = rows(A) @ W^T + bias, rows split across A0|A1 (concat).
// mode 0/1: write [B][H][nrows][64]; mode 2: write transposed [B][H][64][nrows].
// ---------------------------------------------------------------------------
__global__ __launch_bounds__(256) void proj_qkv_kernel(
    const __bf16* __restrict__ A0, long a0_bs,
    const __bf16* __restrict__ A1, long a1_bs, int n0, int nrows,
    const __bf16* __restrict__ W, const float* __restrict__ bias,
    __bf16* __restrict__ out, int mode) {
  const int wave = threadIdx.x >> 5, lane = threadIdx.x & 31;
  const int b    = blockIdx.z;
  const int row0 = blockIdx.x * 128 + wave * 16;
  const int nb   = blockIdx.y * 64;
  const int r    = lane & 15;
  const int kbA  = (lane & 16) ? 8 : 0;
  const int kbB  = (lane & 16) ? 8 : 0;

  const int arow = row0 + r;
  const __bf16* ap =
      (arow < n0) ? (A0 + (size_t)b * a0_bs + (size_t)arow * E_SZ)
                  : (A1 + (size_t)b * a1_bs + (size_t)(arow - n0) * E_SZ);
  const __bf16* wp = W + (size_t)(nb + r) * E_SZ;

  v8f acc[4] = {};
  for (int k = 0; k < E_SZ; k += 32) {
    v16bf af = fragA_ld(ap + k + kbA);
#pragma unroll
    for (int nt = 0; nt < 4; ++nt) {
      v16bf bf = fragB_ld(wp + (size_t)nt * 16 * E_SZ + k + kbB);
      acc[nt] = wmma_bf16(af, bf, acc[nt]);
    }
  }

  const int mhi = (lane & 16) ? 8 : 0;
  const size_t out_b = (size_t)b * H_SZ * (size_t)nrows * D_SZ;
#pragma unroll
  for (int nt = 0; nt < 4; ++nt) {
    const int n = nb + nt * 16 + r;
    const int h = n >> 6, d = n & 63;
    const float bv = bias[n];
#pragma unroll
    for (int rr = 0; rr < 8; ++rr) {
      const int row = row0 + rr + mhi;
      const float v = acc[nt][rr] + bv;
      size_t idx;
      if (mode == 2)
        idx = out_b + ((size_t)h * D_SZ + d) * (size_t)nrows + row;  // V^T
      else
        idx = out_b + ((size_t)h * nrows + row) * D_SZ + d;          // Q / K
      out[idx] = (__bf16)v;
    }
  }
}

// ---------------------------------------------------------------------------
// Flash attention with async double-buffered K/V^T staging in LDS.
// Block = 8 waves sharing one (b,h); each wave owns a 16-query strip.
// Per 64-key block: 256 threads issue 4x global_load_async_to_lds_b128
// (K tile 8KB + V^T tile 8KB) for the NEXT block while computing the current
// one from LDS; s_wait_asynccnt + barriers sequence the buffers.
// ---------------------------------------------------------------------------
__global__ __launch_bounds__(256) void flash_attn_kernel(
    const __bf16* __restrict__ Q, const __bf16* __restrict__ K,
    const __bf16* __restrict__ VT, __bf16* __restrict__ CTX) {
  __shared__ __bf16 kbuf[2][64 * 64];  // [key][d], 8KB per buffer
  __shared__ __bf16 vbuf[2][64 * 64];  // [d][key], 8KB per buffer
  __shared__ __bf16 plds[8][16 * 72];  // per-wave P tile 16x64, stride 72

  const int t    = threadIdx.x;
  const int wave = t >> 5, lane = t & 31;
  const int b = blockIdx.z, h = blockIdx.y;
  const int q0 = blockIdx.x * 128 + wave * 16;
  const size_t bh = (size_t)b * H_SZ + h;
  const __bf16* qp = Q + bh * (size_t)SQ_SZ * D_SZ;
  const __bf16* kp = K + bh * (size_t)SKV_SZ * D_SZ;
  const __bf16* vp = VT + bh * (size_t)D_SZ * SKV_SZ;
  const int r   = lane & 15;
  const int kbA = (lane & 16) ? 8 : 0;
  const int kbB = (lane & 16) ? 8 : 0;
  const int mhi = (lane & 16) ? 8 : 0;

  // Per-thread slot for cooperative tile staging: 64 rows x 128B rows.
  const int srow = t >> 3;          // 0..31
  const int scol = (t & 7) * 8;     // element offset, 16B granules

  const __bf16* qrow = qp + (size_t)(q0 + r) * D_SZ + kbA;
  const v16bf qf0 = fragA_ld(qrow);
  const v16bf qf1 = fragA_ld(qrow + 32);

  v8f o0 = {}, o1 = {}, o2 = {}, o3 = {};
  float mrow[8], lrow[8];
#pragma unroll
  for (int rr = 0; rr < 8; ++rr) { mrow[rr] = -3.0e38f; lrow[rr] = 0.0f; }
  __bf16* pl = &plds[wave][0];

  // Issue async loads of the 64-key block starting at k0 into buffer `buf`.
  auto issue = [&](int buf, int k0) {
#pragma unroll
    for (int s = 0; s < 2; ++s) {
      const int row = srow + s * 32;
      async_ld_b128((lds_bf16*)&kbuf[buf][row * 64 + scol],
                    kp + (size_t)(k0 + row) * D_SZ + scol);
      async_ld_b128((lds_bf16*)&vbuf[buf][row * 64 + scol],
                    vp + (size_t)row * SKV_SZ + k0 + scol);
    }
  };

  issue(0, 0);
  for (int k0 = 0; k0 < SKV_SZ; k0 += 64) {
    const int cur = (k0 >> 6) & 1;
    if (k0 + 64 < SKV_SZ) {
      issue(cur ^ 1, k0 + 64);
      asm volatile("s_wait_asynccnt 0x4" ::: "memory");  // current tiles landed
    } else {
      asm volatile("s_wait_asynccnt 0x0" ::: "memory");
    }
    __syncthreads();  // publish staged tiles to all waves

    const __bf16* kl = &kbuf[cur][0];
    const __bf16* vl = &vbuf[cur][0];

    // Scores: 4 N-tiles x (K=64 -> 2 WMMA steps)
    v8f s0 = {}, s1 = {}, s2 = {}, s3 = {};
    {
      const __bf16* k0p = kl + (0 * 16 + r) * 64 + kbB;
      const __bf16* k1p = kl + (1 * 16 + r) * 64 + kbB;
      const __bf16* k2p = kl + (2 * 16 + r) * 64 + kbB;
      const __bf16* k3p = kl + (3 * 16 + r) * 64 + kbB;
      s0 = wmma_bf16(qf0, fragB_ld(k0p), s0);
      s0 = wmma_bf16(qf1, fragB_ld(k0p + 32), s0);
      s1 = wmma_bf16(qf0, fragB_ld(k1p), s1);
      s1 = wmma_bf16(qf1, fragB_ld(k1p + 32), s1);
      s2 = wmma_bf16(qf0, fragB_ld(k2p), s2);
      s2 = wmma_bf16(qf1, fragB_ld(k2p + 32), s2);
      s3 = wmma_bf16(qf0, fragB_ld(k3p), s3);
      s3 = wmma_bf16(qf1, fragB_ld(k3p + 32), s3);
    }

    // Online softmax across the 64-key block (16-lane butterfly reductions).
#pragma unroll
    for (int rr = 0; rr < 8; ++rr) {
      const float a0 = s0[rr] * 0.125f;  // 1/sqrt(64)
      const float a1 = s1[rr] * 0.125f;
      const float a2 = s2[rr] * 0.125f;
      const float a3 = s3[rr] * 0.125f;
      float mx = fmaxf(fmaxf(a0, a1), fmaxf(a2, a3));
#pragma unroll
      for (int msk = 1; msk <= 8; msk <<= 1)
        mx = fmaxf(mx, __shfl_xor(mx, msk, 32));
      const float mn = fmaxf(mrow[rr], mx);
      const float p0 = __expf(a0 - mn);
      const float p1 = __expf(a1 - mn);
      const float p2 = __expf(a2 - mn);
      const float p3 = __expf(a3 - mn);
      float sm = (p0 + p1) + (p2 + p3);
#pragma unroll
      for (int msk = 1; msk <= 8; msk <<= 1) sm += __shfl_xor(sm, msk, 32);
      const float alpha = __expf(mrow[rr] - mn);
      lrow[rr] = lrow[rr] * alpha + sm;
      mrow[rr] = mn;
      o0[rr] *= alpha; o1[rr] *= alpha; o2[rr] *= alpha; o3[rr] *= alpha;
      __bf16* pr = pl + (rr + mhi) * 72;  // C-layout row -> A-layout LDS tile
      pr[r]      = (__bf16)p0;
      pr[16 + r] = (__bf16)p1;
      pr[32 + r] = (__bf16)p2;
      pr[48 + r] = (__bf16)p3;
    }

    // P (16x64) as two A fragments (same-wave LDS: ordered, no barrier).
    const v16bf pf0 = fragA_ld(pl + r * 72 + kbA);
    const v16bf pf1 = fragA_ld(pl + r * 72 + 32 + kbA);

    // O += P @ V-block : 4 d-tiles x 2 key-chunks
    {
      const __bf16* v0p = vl + (0 * 16 + r) * 64 + kbB;
      const __bf16* v1p = vl + (1 * 16 + r) * 64 + kbB;
      const __bf16* v2p = vl + (2 * 16 + r) * 64 + kbB;
      const __bf16* v3p = vl + (3 * 16 + r) * 64 + kbB;
      o0 = wmma_bf16(pf0, fragB_ld(v0p), o0);
      o0 = wmma_bf16(pf1, fragB_ld(v0p + 32), o0);
      o1 = wmma_bf16(pf0, fragB_ld(v1p), o1);
      o1 = wmma_bf16(pf1, fragB_ld(v1p + 32), o1);
      o2 = wmma_bf16(pf0, fragB_ld(v2p), o2);
      o2 = wmma_bf16(pf1, fragB_ld(v2p + 32), o2);
      o3 = wmma_bf16(pf0, fragB_ld(v3p), o3);
      o3 = wmma_bf16(pf1, fragB_ld(v3p + 32), o3);
    }
    __syncthreads();  // all waves done reading before next issue overwrites
  }

#pragma unroll
  for (int rr = 0; rr < 8; ++rr) {
    const float inv = 1.0f / lrow[rr];
    const size_t row = (size_t)b * SQ_SZ + q0 + rr + mhi;
    __bf16* cp = CTX + row * E_SZ + h * D_SZ + r;
    cp[0]  = (__bf16)(o0[rr] * inv);
    cp[16] = (__bf16)(o1[rr] * inv);
    cp[32] = (__bf16)(o2[rr] * inv);
    cp[48] = (__bf16)(o3[rr] * inv);
  }
}

// ---------------------------------------------------------------------------
// Output projection: out(f32) = ctx(bf16) @ Wo^T + bo
// ---------------------------------------------------------------------------
__global__ __launch_bounds__(256) void out_proj_kernel(
    const __bf16* __restrict__ CTX, const __bf16* __restrict__ W,
    const float* __restrict__ bias, float* __restrict__ out) {
  const int wave = threadIdx.x >> 5, lane = threadIdx.x & 31;
  const int row0 = blockIdx.x * 128 + wave * 16;  // rows over B*SQ
  const int nb   = blockIdx.y * 64;
  const int r    = lane & 15;
  const int kbA  = (lane & 16) ? 8 : 0;
  const int kbB  = (lane & 16) ? 8 : 0;
  const __bf16* ap = CTX + (size_t)(row0 + r) * E_SZ;

  v8f acc[4] = {};
  for (int k = 0; k < E_SZ; k += 32) {
    v16bf af = fragA_ld(ap + k + kbA);
#pragma unroll
    for (int nt = 0; nt < 4; ++nt)
      acc[nt] = wmma_bf16(
          af, fragB_ld(W + (size_t)(nb + nt * 16 + r) * E_SZ + k + kbB),
          acc[nt]);
  }
  const int mhi = (lane & 16) ? 8 : 0;
#pragma unroll
  for (int nt = 0; nt < 4; ++nt) {
    const int n = nb + nt * 16 + r;
    const float bv = bias[n];
#pragma unroll
    for (int rr = 0; rr < 8; ++rr)
      out[(size_t)(row0 + rr + mhi) * E_SZ + n] = acc[nt][rr] + bv;
  }
}

// ---------------------------------------------------------------------------
extern "C" void kernel_launch(void* const* d_in, const int* in_sizes, int n_in,
                              void* d_out, int out_size, void* d_ws,
                              size_t ws_size, hipStream_t stream) {
  const float* x     = (const float*)d_in[0];
  const float* cache = (const float*)d_in[1];
  const float* ipw   = (const float*)d_in[2];
  const float* ipb   = (const float*)d_in[3];
  const float* opw   = (const float*)d_in[4];
  const float* opb   = (const float*)d_in[5];
  float* out = (float*)d_out;

  size_t off = 0;
  auto carve = [&](size_t elems) -> __bf16* {
    __bf16* p = (__bf16*)((char*)d_ws + off);
    off += ((elems * sizeof(__bf16) + 255) & ~(size_t)255);
    return p;
  };
  __bf16* xb   = carve((size_t)B_SZ * SQ_SZ * E_SZ);          // x bf16
  __bf16* cb   = carve((size_t)B_SZ * SC_SZ * E_SZ);          // cache bf16
  __bf16* wqkv = carve((size_t)3 * E_SZ * E_SZ);              // in_proj_w bf16
  __bf16* wo   = carve((size_t)E_SZ * E_SZ);                  // out_proj_w bf16
  __bf16* qb   = carve((size_t)B_SZ * H_SZ * SQ_SZ * D_SZ);   // Q  [B,H,SQ,64]
  __bf16* kb   = carve((size_t)B_SZ * H_SZ * SKV_SZ * D_SZ);  // K  [B,H,SKV,64]
  __bf16* vtb  = carve((size_t)B_SZ * H_SZ * D_SZ * SKV_SZ);  // V^T[B,H,64,SKV]
  __bf16* ctx  = carve((size_t)B_SZ * SQ_SZ * E_SZ);          // ctx bf16

  auto cvt = [&](const float* s, __bf16* d, int n) {
    cvt_bf16_kernel<<<dim3((n / 4 + 255) / 256), dim3(256), 0, stream>>>(s, d, n);
  };
  cvt(x, xb, B_SZ * SQ_SZ * E_SZ);
  cvt(cache, cb, B_SZ * SC_SZ * E_SZ);
  cvt(ipw, wqkv, 3 * E_SZ * E_SZ);
  cvt(opw, wo, E_SZ * E_SZ);

  // Q = x @ Wq^T + bq
  proj_qkv_kernel<<<dim3(SQ_SZ / 128, E_SZ / 64, B_SZ), 256, 0, stream>>>(
      xb, (long)SQ_SZ * E_SZ, xb, (long)SQ_SZ * E_SZ, SQ_SZ, SQ_SZ, wqkv, ipb,
      qb, 0);
  // K = [cache;x] @ Wk^T + bk
  proj_qkv_kernel<<<dim3(SKV_SZ / 128, E_SZ / 64, B_SZ), 256, 0, stream>>>(
      cb, (long)SC_SZ * E_SZ, xb, (long)SQ_SZ * E_SZ, SC_SZ, SKV_SZ,
      wqkv + (size_t)E_SZ * E_SZ, ipb + E_SZ, kb, 1);
  // V^T = ([cache;x] @ Wv^T + bv)^T per head
  proj_qkv_kernel<<<dim3(SKV_SZ / 128, E_SZ / 64, B_SZ), 256, 0, stream>>>(
      cb, (long)SC_SZ * E_SZ, xb, (long)SQ_SZ * E_SZ, SC_SZ, SKV_SZ,
      wqkv + (size_t)2 * E_SZ * E_SZ, ipb + 2 * E_SZ, vtb, 2);

  flash_attn_kernel<<<dim3(SQ_SZ / 128, H_SZ, B_SZ), 256, 0, stream>>>(qb, kb,
                                                                       vtb, ctx);

  out_proj_kernel<<<dim3((B_SZ * SQ_SZ) / 128, E_SZ / 64, 1), 256, 0, stream>>>(
      ctx, wo, opb, out);
}
